// Model_26070451487322
// MI455X (gfx1250) — matrix operand order repaired and checked by reading, not compiled, not run on hardware
//
#include <hip/hip_runtime.h>
#include <stdint.h>

// ---------------- problem constants ----------------
#define NN      30000     // nodes
#define EE      480000    // edges
#define RR      8192      // rows
#define NCOL    128
#define NIN     128
#define EIN     64
#define NHID    256
#define EHID    128
#define NOUT    256
#define EOUT    64
#define NCLASS  10

typedef __bf16 bf16_t;
typedef __attribute__((ext_vector_type(16))) __bf16 v16bf;
typedef __attribute__((ext_vector_type(8)))  __bf16 v8bf;
typedef __attribute__((ext_vector_type(4)))  __bf16 v4bf;
typedef __attribute__((ext_vector_type(4)))  float  v4f_;
typedef __attribute__((ext_vector_type(8)))  float  v8f;

__device__ __forceinline__ float leakyf(float x) { return x > 0.f ? x : 0.01f * x; }

// ---------------- CDNA5 async copy (global -> LDS, ASYNCcnt-tracked) ----------------
__device__ __forceinline__ void async_b128(uint32_t lds_off, const void* gaddr) {
  asm volatile("global_load_async_to_lds_b128 %0, %1, off" :: "v"(lds_off), "v"(gaddr) : "memory");
}
__device__ __forceinline__ void wait_async0() {
  asm volatile("s_wait_asynccnt 0" ::: "memory");
}
__device__ __forceinline__ uint32_t lds_off(const void* p) {
  return (uint32_t)(uintptr_t)p;  // low 32 bits of a flat shared pointer = LDS offset
}

// ---------------- WMMA fragment helpers (bf16 A/B, f32 acc, 16x16x32) ----------------
// A-matrix 16x32 bf16 (ISA 7.12.2): lane m = lane&15, half q = (lane>>4)*8.
// element[h*8+j] = A[m][h*16 + q + j]  -> two contiguous 16B runs per lane.
__device__ __forceinline__ v16bf a_frag_lds(const bf16_t* tile, int ldk, int kbase, int lane) {
  int m = lane & 15;
  int q = (lane >> 4) << 3;
  const bf16_t* row = tile + m * ldk + kbase;
  v8bf lo = *(const v8bf*)(row + q);
  v8bf hi = *(const v8bf*)(row + 16 + q);
  return __builtin_shufflevector(lo, hi, 0, 1, 2, 3, 4, 5, 6, 7, 8, 9, 10, 11, 12, 13, 14, 15);
}

// B fragments are prepacked: P[(frag*32 + lane)*16 + e] -> one 32B vector load per lane.
__device__ __forceinline__ v16bf b_frag_packed(const bf16_t* P, int frag, int lane) {
  return *(const v16bf*)(P + ((size_t)frag * 32 + lane) * 16);
}

__device__ __forceinline__ v8f wmma_bf16(v16bf a, v16bf b, v8f c) {
  return __builtin_amdgcn_wmma_f32_16x16x32_bf16(false, a, false, b, (short)0, c, false, false);
}

// f32x4 -> bf16x4 convert+store (8B LDS store)
__device__ __forceinline__ void cvt_store4(bf16_t* dst, const float* src) {
  v4f_ t = *(const v4f_*)src;
  v4bf o;
  o[0] = (bf16_t)t[0]; o[1] = (bf16_t)t[1]; o[2] = (bf16_t)t[2]; o[3] = (bf16_t)t[3];
  *(v4bf*)dst = o;
}
__device__ __forceinline__ void cvt_store4s(bf16_t* dst, const float* src, float s) {
  v4f_ t = *(const v4f_*)src;
  v4bf o;
  o[0] = (bf16_t)(t[0] * s); o[1] = (bf16_t)(t[1] * s);
  o[2] = (bf16_t)(t[2] * s); o[3] = (bf16_t)(t[3] * s);
  *(v4bf*)dst = o;
}

// ---------------- utility kernels ----------------
__global__ void k_zero(float* p, int n) {
  int i = blockIdx.x * blockDim.x + threadIdx.x;
  int s = gridDim.x * blockDim.x;
  for (; i < n; i += s) p[i] = 0.f;
}

// Pack W[K][NO] (f32) into WMMA B-fragment order:
// P[((kt*(NO/16)+nt)*32 + lane)*16 + e] -> W[kt*32 + (e>>3)*16 + (lane>>4)*8 + (e&7)][nt*16 + (lane&15)]
__global__ void k_pack_b(const float* __restrict__ W, bf16_t* __restrict__ P, int K, int NO) {
  int idx = blockIdx.x * blockDim.x + threadIdx.x;
  if (idx >= K * NO) return;
  int e    = idx & 15;
  int lane = (idx >> 4) & 31;
  int frag = idx >> 9;
  int NT   = NO >> 4;
  int nt   = frag % NT;
  int kt   = frag / NT;
  int k = kt * 32 + ((e >> 3) << 4) + ((lane >> 4) << 3) + (e & 7);      // table order, matches A fragment
  int n = nt * 16 + (lane & 15);
  P[idx] = (bf16_t)W[(size_t)k * NO + n];
}

__global__ void k_deg(const int* __restrict__ dst, float* deg, int e) {
  int i = blockIdx.x * blockDim.x + threadIdx.x;
  if (i < e) atomicAdd(&deg[dst[i]], 1.0f);
}

// ---------------- ESAGE edge GEMM: e_out = leaky([nodeF[src]; edgeF] @ W + b) ----------------
// Fused segment-sum: atomicAdd into hsum[dst]. eout may be null (layer 2).
template <typename TN, typename TE, int KA, int KB, int NT>
__global__ __launch_bounds__(128)
void k_edge(const TN* __restrict__ nodeF, const TE* __restrict__ edgeF,
            const int* __restrict__ src, const int* __restrict__ dst,
            const bf16_t* __restrict__ P, const float* __restrict__ bias,
            bf16_t* __restrict__ eout, float* __restrict__ hsum) {
  constexpr int K  = KA + KB;
  constexpr int KT = K / 32;
  constexpr int NO = NT * 16;
  __shared__ __align__(64) bf16_t tile[4][16][K];
  int wave = threadIdx.x >> 5, lane = threadIdx.x & 31;
  int edgeBase = (blockIdx.x * 4 + wave) * 16;

  // ---- stage gathered A tile (concat rows) into LDS as bf16 ----
  if constexpr (sizeof(TN) == 2) {
    // bf16 source: async DMA gathered node rows (KA*2 bytes each)
    for (int r = 0; r < 16; ++r) {
      int sn = src[edgeBase + r];
      const char* g = (const char*)(nodeF + (size_t)sn * KA);
      uint32_t l = lds_off(&tile[wave][r][0]);
      for (int c = lane * 16; c < KA * 2; c += 512) async_b128(l + c, g + c);
    }
  } else {
    for (int r = 0; r < 16; ++r) {
      int sn = src[edgeBase + r];
      const TN* nrow = nodeF + (size_t)sn * KA;
      for (int c = lane * 4; c < KA; c += 128) cvt_store4(&tile[wave][r][c], (const float*)nrow + c);
    }
  }
  if constexpr (sizeof(TE) == 2) {
    // bf16 edge rows are contiguous in global: async DMA 16*KB*2 bytes
    constexpr int ROWB = KB * 2;
    const char* g = (const char*)(edgeF + (size_t)edgeBase * KB);
    for (int t = lane * 16; t < 16 * ROWB; t += 512) {
      int r = t / ROWB, cB = t % ROWB;
      async_b128(lds_off(&tile[wave][r][KA]) + cB, g + t);
    }
  } else {
    for (int r = 0; r < 16; ++r) {
      const TE* erow = edgeF + (size_t)(edgeBase + r) * KB;
      for (int c = lane * 4; c < KB; c += 128) cvt_store4(&tile[wave][r][KA + c], (const float*)erow + c);
    }
  }
  if constexpr (sizeof(TN) == 2 || sizeof(TE) == 2) wait_async0();
  __syncthreads();

  v8f acc[NT];
  v8f zero = {0.f, 0.f, 0.f, 0.f, 0.f, 0.f, 0.f, 0.f};
#pragma unroll
  for (int n = 0; n < NT; ++n) acc[n] = zero;

  for (int kt = 0; kt < KT; ++kt) {
    if (kt + 1 < KT) __builtin_prefetch(P + (size_t)(kt + 1) * NT * 512, 0, 0);
    v16bf a = a_frag_lds(&tile[wave][0][0], K, kt * 32, lane);
#pragma unroll
    for (int n = 0; n < NT; ++n) {
      v16bf b = b_frag_packed(P, kt * NT + n, lane);
      acc[n] = wmma_bf16(a, b, acc[n]);
    }
  }

  int ncol = lane & 15;
  int rb   = (lane >> 4) << 3;
  float bs[NT];
#pragma unroll
  for (int n = 0; n < NT; ++n) bs[n] = bias[n * 16 + ncol];
#pragma unroll
  for (int v = 0; v < 8; ++v) {
    int eidx = edgeBase + rb + v;
    int d = dst[eidx];
    float* hrow = hsum + (size_t)d * NO;
#pragma unroll
    for (int n = 0; n < NT; ++n) {
      float val = leakyf(acc[n][v] + bs[n]);
      if (eout) eout[(size_t)eidx * NO + n * 16 + ncol] = (bf16_t)val;
      atomicAdd(&hrow[n * 16 + ncol], val);
    }
  }
}

// ---------------- ESAGE node GEMM: x_out = leaky([f1; hsum/deg] @ W + b) ----------------
template <typename T1, int KA, int KB, int NTOT>
__global__ __launch_bounds__(128)
void k_node(const T1* __restrict__ f1, const float* __restrict__ hsum,
            const float* __restrict__ deg, const bf16_t* __restrict__ P,
            const float* __restrict__ bias, int nnode,
            float* __restrict__ xf_out, bf16_t* __restrict__ xb_out) {
  constexpr int K  = KA + KB;
  constexpr int KT = K / 32;
  constexpr int NT = 8;  // 128 columns per launch; gridDim.y selects column half
  __shared__ __align__(64) bf16_t tile[4][16][K];
  int wave = threadIdx.x >> 5, lane = threadIdx.x & 31;
  int rowBase = (blockIdx.x * 4 + wave) * 16;
  int nbase   = blockIdx.y * 128;
  bool used_async = false;

  for (int r = 0; r < 16; ++r) {
    int node = rowBase + r;
    if (node < nnode) {
      float dv = deg[node];
      float inv = 1.f / (dv > 1.f ? dv : 1.f);
      if constexpr (sizeof(T1) == 2) {
        const char* g = (const char*)(f1 + (size_t)node * KA);
        uint32_t l = lds_off(&tile[wave][r][0]);
        for (int c = lane * 16; c < KA * 2; c += 512) async_b128(l + c, g + c);
        used_async = true;
      } else {
        const T1* frow = f1 + (size_t)node * KA;
        for (int c = lane * 4; c < KA; c += 128) cvt_store4(&tile[wave][r][c], (const float*)frow + c);
      }
      const float* hrow = hsum + (size_t)node * KB;
      for (int c = lane * 4; c < KB; c += 128) cvt_store4s(&tile[wave][r][KA + c], hrow + c, inv);
    } else {
      v4bf z = {(bf16_t)0.f, (bf16_t)0.f, (bf16_t)0.f, (bf16_t)0.f};
      for (int c = lane * 4; c < K; c += 128) *(v4bf*)&tile[wave][r][c] = z;
    }
  }
  if (used_async) wait_async0();
  __syncthreads();

  v8f acc[NT];
  v8f zero = {0.f, 0.f, 0.f, 0.f, 0.f, 0.f, 0.f, 0.f};
#pragma unroll
  for (int n = 0; n < NT; ++n) acc[n] = zero;

  constexpr int NTW = NTOT / 16;  // fragments per k-row in packed weights
  for (int kt = 0; kt < KT; ++kt) {
    if (kt + 1 < KT) __builtin_prefetch(P + (size_t)(kt + 1) * NTW * 512, 0, 0);
    v16bf a = a_frag_lds(&tile[wave][0][0], K, kt * 32, lane);
#pragma unroll
    for (int n = 0; n < NT; ++n) {
      v16bf b = b_frag_packed(P, kt * NTW + (nbase >> 4) + n, lane);
      acc[n] = wmma_bf16(a, b, acc[n]);
    }
  }

  int ncol = lane & 15;
  int rb   = (lane >> 4) << 3;
#pragma unroll
  for (int v = 0; v < 8; ++v) {
    int node = rowBase + rb + v;
    if (node < nnode) {
#pragma unroll
      for (int n = 0; n < NT; ++n) {
        int col = nbase + n * 16 + ncol;
        float val = leakyf(acc[n][v] + bias[col]);
        if (xf_out) xf_out[(size_t)node * NTOT + col] = val;
        xb_out[(size_t)node * NTOT + col] = (bf16_t)val;
      }
    }
  }
}

// ---------------- rowWr = x2[row_idx] @ Wr  (gathered 8192x256 @ 256x64) ----------------
__global__ __launch_bounds__(128)
void k_rowwr(const bf16_t* __restrict__ x2b, const int* __restrict__ row_idx,
             const bf16_t* __restrict__ P, float* __restrict__ out) {
  __shared__ __align__(64) bf16_t tile[4][16][256];
  int wave = threadIdx.x >> 5, lane = threadIdx.x & 31;
  int rowBase = (blockIdx.x * 4 + wave) * 16;
  for (int r = 0; r < 16; ++r) {
    int node = row_idx[rowBase + r];
    const char* g = (const char*)(x2b + (size_t)node * 256);
    async_b128(lds_off(&tile[wave][r][0]) + lane * 16, g + lane * 16);  // 512B row = 32 lanes x 16B
  }
  wait_async0();
  __syncthreads();

  v8f acc[4];
  v8f zero = {0.f, 0.f, 0.f, 0.f, 0.f, 0.f, 0.f, 0.f};
#pragma unroll
  for (int n = 0; n < 4; ++n) acc[n] = zero;
  for (int kt = 0; kt < 8; ++kt) {
    v16bf a = a_frag_lds(&tile[wave][0][0], 256, kt * 32, lane);
#pragma unroll
    for (int n = 0; n < 4; ++n) {
      v16bf b = b_frag_packed(P, kt * 4 + n, lane);
      acc[n] = wmma_bf16(a, b, acc[n]);
    }
  }
  int ncol = lane & 15, rb = (lane >> 4) << 3;
#pragma unroll
  for (int v = 0; v < 8; ++v)
#pragma unroll
    for (int n = 0; n < 4; ++n)
      out[(size_t)(rowBase + rb + v) * 64 + n * 16 + ncol] = acc[n][v];
}

// ---------------- attention (128 tokens, f32 VALU — negligible FLOPs) ----------------
__global__ void k_att_c(const float* __restrict__ x1f, const int* __restrict__ col_idx,
                        const float* __restrict__ pe, float* __restrict__ c) {
  int i = blockIdx.x, n = threadIdx.x;
  c[i * 256 + n] = x1f[(size_t)col_idx[i] * 256 + n] + pe[i * 256 + n];
}

__global__ void k_att_qkv(const float* __restrict__ c, const float* __restrict__ Wq,
                          const float* __restrict__ Wk, const float* __restrict__ Wv,
                          float* __restrict__ q, float* __restrict__ k, float* __restrict__ v) {
  const float* W = blockIdx.y == 0 ? Wq : (blockIdx.y == 1 ? Wk : Wv);
  float* o       = blockIdx.y == 0 ? q  : (blockIdx.y == 1 ? k  : v);
  int i = blockIdx.x, n = threadIdx.x;
  float s = 0.f;
  for (int t = 0; t < 256; ++t) s += c[i * 256 + t] * W[t * 256 + n];
  o[i * 256 + n] = s;
}

__global__ void k_att_soft(const float* __restrict__ q, const float* __restrict__ k,
                           float* __restrict__ dist) {
  __shared__ float red[128];
  int i = blockIdx.x, j = threadIdx.x;
  float s = 0.f;
  for (int t = 0; t < 256; ++t) s += q[i * 256 + t] * k[j * 256 + t];
  s *= 0.0625f;  // 1/sqrt(256)
  red[j] = s; __syncthreads();
  for (int st = 64; st > 0; st >>= 1) { if (j < st) red[j] = fmaxf(red[j], red[j + st]); __syncthreads(); }
  float m = red[0]; __syncthreads();
  float e = __expf(s - m);
  red[j] = e; __syncthreads();
  for (int st = 64; st > 0; st >>= 1) { if (j < st) red[j] += red[j + st]; __syncthreads(); }
  dist[i * 128 + j] = e / red[0];
}

__global__ void k_att_av(const float* __restrict__ dist, const float* __restrict__ v,
                         float* __restrict__ c2) {
  int i = blockIdx.x, n = threadIdx.x;
  float s = 0.f;
  for (int j = 0; j < 128; ++j) s += dist[i * 128 + j] * v[j * 256 + n];
  c2[i * 256 + n] = 2.0f * s;  // dist@v + dist@v
}

__global__ void k_att_ref(const float* __restrict__ c2, const float* __restrict__ Wref,
                          const float* __restrict__ bref, float* __restrict__ nc) {
  int i = blockIdx.x, n = threadIdx.x;
  float s = bref[n];
  for (int t = 0; t < 256; ++t) s += c2[i * 256 + t] * Wref[t * 256 + n];
  nc[i * 256 + n] = 1.f / (1.f + __expf(-s));
}

__global__ void k_colwc(const float* __restrict__ nc, const float* __restrict__ Wc,
                        float* __restrict__ colWc) {
  int j = blockIdx.x, t = threadIdx.x;  // 128 x 64
  float s = 0.f;
  for (int k = 0; k < 256; ++k) s += nc[j * 256 + k] * Wc[k * 64 + t];
  colWc[j * 64 + t] = s;
}

// ---------------- pred = leaky(rowWr[r] + colWc[j] + bep)  (268 MB output) ----------------
__global__ __launch_bounds__(256)
void k_pred(const float* __restrict__ rowWr, const float* __restrict__ colWc,
            const float* __restrict__ bep, float* __restrict__ pred) {
  long long idx = (long long)blockIdx.x * 256 + threadIdx.x;
  int t = (int)(idx & 63);
  long long pair = idx >> 6;
  int j = (int)(pair & 127);
  long long r = pair >> 7;
  float val = rowWr[r * 64 + t] + colWc[j * 64 + t] + bep[t];
  pred[idx] = leakyf(val);
}

// ---------------- classifier: masked imputation -> Linear(8192,10) -> log_softmax ----------------
__global__ __launch_bounds__(256)
void k_cls(const float* __restrict__ pred, const float* __restrict__ realv,
           const unsigned char* __restrict__ mask, const float* __restrict__ Wcls,
           const float* __restrict__ bcls, float* __restrict__ res) {
  __shared__ float red[256][NCLASS];
  int r = blockIdx.x, tid = threadIdx.x;
  float a[NCLASS];
#pragma unroll
  for (int c = 0; c < NCLASS; ++c) a[c] = 0.f;
  const float* prow = pred  + (size_t)r * 8192;
  const float* rrow = realv + (size_t)r * 8192;
  const unsigned char* mrow = mask + (size_t)r * NCOL;
  for (int k = tid; k < 8192; k += 256) {
    int j = k >> 6;
    float val = mrow[j] ? rrow[k] : prow[k];
    const float* w = Wcls + (size_t)k * NCLASS;
#pragma unroll
    for (int c = 0; c < NCLASS; ++c) a[c] += val * w[c];
  }
#pragma unroll
  for (int c = 0; c < NCLASS; ++c) red[tid][c] = a[c];
  __syncthreads();
  for (int st = 128; st > 0; st >>= 1) {
    if (tid < st)
#pragma unroll
      for (int c = 0; c < NCLASS; ++c) red[tid][c] += red[tid + st][c];
    __syncthreads();
  }
  if (tid == 0) {
    float l[NCLASS], m = -1e30f;
#pragma unroll
    for (int c = 0; c < NCLASS; ++c) { l[c] = red[0][c] + bcls[c]; m = fmaxf(m, l[c]); }
    float s = 0.f;
#pragma unroll
    for (int c = 0; c < NCLASS; ++c) s += __expf(l[c] - m);
    float lse = m + __logf(s);
#pragma unroll
    for (int c = 0; c < NCLASS; ++c) res[(size_t)r * NCLASS + c] = l[c] - lse;
  }
}

// ---------------- host launcher ----------------
extern "C" void kernel_launch(void* const* d_in, const int* in_sizes, int n_in,
                              void* d_out, int out_size, void* d_ws, size_t ws_size,
                              hipStream_t stream) {
  (void)in_sizes; (void)n_in; (void)out_size; (void)ws_size;
  const float* x       = (const float*)d_in[0];
  const float* efeat   = (const float*)d_in[1];
  const int*   src     = (const int*)d_in[2];
  const int*   dst     = (const int*)d_in[3];
  const int*   row_idx = (const int*)d_in[4];
  const int*   col_idx = (const int*)d_in[5];
  const float* realv   = (const float*)d_in[6];
  const unsigned char* mask = (const unsigned char*)d_in[7];
  const float* W1e = (const float*)d_in[8];   const float* b1e = (const float*)d_in[9];
  const float* W1n = (const float*)d_in[10];  const float* b1n = (const float*)d_in[11];
  const float* W2e = (const float*)d_in[12];  const float* b2e = (const float*)d_in[13];
  const float* W2n = (const float*)d_in[14];  const float* b2n = (const float*)d_in[15];
  const float* pe  = (const float*)d_in[16];
  const float* Wq  = (const float*)d_in[17];
  const float* Wk  = (const float*)d_in[18];
  const float* Wv  = (const float*)d_in[19];
  const float* Wref= (const float*)d_in[20];  const float* bref= (const float*)d_in[21];
  const float* Wr  = (const float*)d_in[22];
  const float* Wc  = (const float*)d_in[23];
  const float* bep = (const float*)d_in[24];
  const float* Wcls= (const float*)d_in[25];  const float* bcls= (const float*)d_in[26];

  char* ws = (char*)d_ws;
  size_t o = 0;
  auto wsalloc = [&](size_t bytes) { void* p = ws + o; o = (o + bytes + 255) & ~(size_t)255; return p; };

  bf16_t* w1e_p = (bf16_t*)wsalloc((size_t)192 * 128 * 2);
  bf16_t* w1n_p = (bf16_t*)wsalloc((size_t)256 * 256 * 2);
  bf16_t* w2e_p = (bf16_t*)wsalloc((size_t)384 * 64 * 2);
  bf16_t* w2n_p = (bf16_t*)wsalloc((size_t)320 * 256 * 2);
  bf16_t* wr_p  = (bf16_t*)wsalloc((size_t)256 * 64 * 2);
  bf16_t* e1b   = (bf16_t*)wsalloc((size_t)EE * EHID * 2);
  float*  x1f   = (float*) wsalloc((size_t)NN * NHID * 4);
  bf16_t* x1b   = (bf16_t*)wsalloc((size_t)NN * NHID * 2);
  bf16_t* x2b   = (bf16_t*)wsalloc((size_t)NN * NOUT * 2);
  float*  hsum1 = (float*) wsalloc((size_t)NN * EHID * 4);   // contiguous zero region start
  float*  hsum2 = (float*) wsalloc((size_t)NN * EOUT * 4);
  float*  deg   = (float*) wsalloc((size_t)NN * 4);
  float*  attc  = (float*) wsalloc((size_t)128 * 256 * 4);
  float*  attq  = (float*) wsalloc((size_t)128 * 256 * 4);
  float*  attk  = (float*) wsalloc((size_t)128 * 256 * 4);
  float*  attv  = (float*) wsalloc((size_t)128 * 256 * 4);
  float*  distm = (float*) wsalloc((size_t)128 * 128 * 4);
  float*  c2    = (float*) wsalloc((size_t)128 * 256 * 4);
  float*  ncol  = (float*) wsalloc((size_t)128 * 256 * 4);
  float*  colWc = (float*) wsalloc((size_t)128 * 64 * 4);
  float*  rowWr = (float*) wsalloc((size_t)RR * 64 * 4);

  float* res_out  = (float*)d_out;
  float* pred_out = res_out + (size_t)RR * NCLASS;

  // zero segment-sum accumulators + degree counts (contiguous)
  int nz = NN * EHID + NN * EOUT + NN;
  k_zero<<<4096, 256, 0, stream>>>(hsum1, nz);

  // pack weights into WMMA B-fragment layout (f32 -> bf16)
  k_pack_b<<<96, 256, 0, stream>>>(W1e, w1e_p, 192, 128);
  k_pack_b<<<256, 256, 0, stream>>>(W1n, w1n_p, 256, 256);
  k_pack_b<<<96, 256, 0, stream>>>(W2e, w2e_p, 384, 64);
  k_pack_b<<<320, 256, 0, stream>>>(W2n, w2n_p, 320, 256);
  k_pack_b<<<64, 256, 0, stream>>>(Wr, wr_p, 256, 64);

  // in-degree counts (shared by both layers)
  k_deg<<<(EE + 255) / 256, 256, 0, stream>>>(dst, deg, EE);

  // --- ESAGE layer 1 ---
  k_edge<float, float, NIN, EIN, EHID / 16>
      <<<EE / 64, 128, 0, stream>>>(x, efeat, src, dst, w1e_p, b1e, e1b, hsum1);
  k_node<float, NIN, EHID, NHID>
      <<<dim3((NN + 63) / 64, 2), 128, 0, stream>>>(x, hsum1, deg, w1n_p, b1n, NN, x1f, x1b);

  // --- ESAGE layer 2 (e2 fully fused into the segment-sum; async-staged A tiles) ---
  k_edge<bf16_t, bf16_t, NHID, EHID, EOUT / 16>
      <<<EE / 64, 128, 0, stream>>>(x1b, e1b, src, dst, w2e_p, b2e, (bf16_t*)nullptr, hsum2);
  k_node<bf16_t, NHID, EOUT, NOUT>
      <<<dim3((NN + 63) / 64, 2), 128, 0, stream>>>(x1b, hsum2, deg, w2n_p, b2n, NN,
                                                    (float*)nullptr, x2b);

  // --- column self-attention (128 tokens) ---
  k_att_c<<<128, 256, 0, stream>>>(x1f, col_idx, pe, attc);
  k_att_qkv<<<dim3(128, 3), 256, 0, stream>>>(attc, Wq, Wk, Wv, attq, attk, attv);
  k_att_soft<<<128, 128, 0, stream>>>(attq, attk, distm);
  k_att_av<<<128, 256, 0, stream>>>(distm, attv, c2);
  k_att_ref<<<128, 256, 0, stream>>>(c2, Wref, bref, ncol);
  k_colwc<<<128, 64, 0, stream>>>(ncol, Wc, colWc);

  // --- row projection (gathered GEMM, async-staged) + edge predictor + classifier ---
  k_rowwr<<<RR / 64, 128, 0, stream>>>(x2b, row_idx, wr_p, rowWr);
  k_pred<<<(unsigned)((size_t)RR * NCOL * EOUT / 256), 256, 0, stream>>>(rowWr, colWc, bep, pred_out);
  k_cls<<<RR, 256, 0, stream>>>(pred_out, realv, mask, Wcls, bcls, res_out);
}
